// Exaone4Attention_90658169684503
// MI455X (gfx1250) — matrix operand order repaired and checked by reading, not compiled, not run on hardware
//
#include <hip/hip_runtime.h>

typedef __bf16 bf16_t;
typedef __attribute__((ext_vector_type(8)))  float  v8f;
typedef __attribute__((ext_vector_type(16))) bf16_t v16bf;
typedef __attribute__((ext_vector_type(8)))  bf16_t v8bf;

#define S_LEN 2048
#define NQH   32
#define NKVH  8
#define HDIM  128
#define WIN   1024

// ---------------------------------------------------------------------------
// WMMA fragment helpers (wave32, V_WMMA_F32_16X16X32_BF16)
// A 16x32: lane L holds row M=L%16; elem e -> K=16*half+e.
// B 32x16: lane L holds col N=L%16; lanes 0-15 K=0..15, lanes 16-31 K=16..31.
// C/D 16x16 f32: lane holds col N=L%16; VGPR i holds row M = i + 8*half.
// ---------------------------------------------------------------------------
__device__ __forceinline__ v16bf make_afrag(const bf16_t* p) {
  v8bf lo = *reinterpret_cast<const v8bf*>(p);
  v8bf hi = *reinterpret_cast<const v8bf*>(p + 8);
  v16bf r;
#pragma unroll
  for (int i = 0; i < 8; ++i) { r[i] = lo[i]; r[i + 8] = hi[i]; }
  return r;
}

__device__ __forceinline__ v16bf make_bfrag(const bf16_t* p) {
  v8bf lo = *reinterpret_cast<const v8bf*>(p);
  v8bf hi = *reinterpret_cast<const v8bf*>(p + 8);
  v16bf r;
#pragma unroll
  for (int i = 0; i < 8; ++i) { r[i] = lo[i]; r[i + 8] = hi[i]; }
  return r;
}

__device__ __forceinline__ v8f wmma_bf16(v16bf a, v16bf b, v8f c) {
  return __builtin_amdgcn_wmma_f32_16x16x32_bf16(false, a, false, b, (short)0, c,
                                                 false, false);
}

// ---------------------------------------------------------------------------
// Async global -> LDS 16B copy (ASYNCcnt-tracked, gfx1250).
// LDS destination given per lane by lds_off; global addr per-lane 64-bit (GV mode).
// ---------------------------------------------------------------------------
__device__ __forceinline__ void async_copy16(unsigned lds_off, const void* gaddr) {
  asm volatile("global_load_async_to_lds_b128 %0, %1, off"
               :: "v"(lds_off), "v"(gaddr)
               : "memory");
}

__device__ __forceinline__ void async_wait_all() {
  asm volatile("s_wait_asynccnt 0x0" ::: "memory");
}

// allow the newest batch of 12 copies to remain in flight
__device__ __forceinline__ void async_wait_12() {
  asm volatile("s_wait_asynccnt 0xc" ::: "memory");
}

// ---------------------------------------------------------------------------
// f32 -> bf16 conversion (row-major copy)
// ---------------------------------------------------------------------------
__global__ void cvt_kernel(const float* __restrict__ src, bf16_t* __restrict__ dst,
                           int n) {
  int i = blockIdx.x * blockDim.x + threadIdx.x;
  if (i < n) dst[i] = (bf16_t)src[i];
}

// ---------------------------------------------------------------------------
// f32 [K,N] -> bf16 [N,K] tiled transpose (weights, done once per weight)
// block (32,8), grid (N/32, K/32)
// ---------------------------------------------------------------------------
__launch_bounds__(256)
__global__ void cvt_transpose_kernel(const float* __restrict__ src,
                                     bf16_t* __restrict__ dst, int K, int N) {
  __shared__ float t[32][33];
  const int k0 = blockIdx.y * 32, n0 = blockIdx.x * 32;
  const int tx = threadIdx.x, ty = threadIdx.y;
#pragma unroll
  for (int i = 0; i < 32; i += 8)
    t[ty + i][tx] = src[(size_t)(k0 + ty + i) * N + n0 + tx];
  __syncthreads();
#pragma unroll
  for (int i = 0; i < 32; i += 8)
    dst[(size_t)(n0 + ty + i) * K + k0 + tx] = (bf16_t)t[tx][ty + i];
}

// ---------------------------------------------------------------------------
// Tiled bf16 WMMA GEMM: C[M,N] (f32) = A[M,K] * BT[N,K]^T  (both bf16)
// Block: 256 thr = 8 waves (2x4); tile 128x256, K-step 64.
// Wave: 64x64 = 4 M-frags x 4 N-frags = 16 WMMAs per 32-wide k-step.
// Double-buffered LDS (~110 KB of CDNA5's 320 KB/WGP): batch i+1 streams in
// via the async engine (ASYNCcnt) while batch i feeds the WMMAs; in-order
// async completion lets s_wait_asynccnt 12 drain exactly the older batch.
// Requires M%128==0, N%256==0, K%64==0 (true for all launches here).
// ---------------------------------------------------------------------------
#define GBM 128
#define GBN 256
#define GBK 64
#define KSTR 72  // LDS row stride (bf16 elems): 144B, 16B-aligned, padded

__launch_bounds__(256)
__global__ void gemm_bf16_kernel(const bf16_t* __restrict__ A,
                                 const bf16_t* __restrict__ BT,
                                 float* __restrict__ C, int M, int N, int K) {
  __shared__ bf16_t As[2][GBM][KSTR];  // [buf][m][k]
  __shared__ bf16_t Bs[2][GBN][KSTR];  // [buf][n][k]
  const int tid  = threadIdx.x;
  const int lane = tid & 31;
  const int wid  = tid >> 5;
  const int half = lane >> 4, l16 = lane & 15;
  const int wm = wid >> 2, wn = wid & 3;
  const int m0 = blockIdx.y * GBM, n0 = blockIdx.x * GBN;

  // 12 async 16B copies per thread fill one (A,B) tile pair
  auto stage = [&](int buf, int kk) {
#pragma unroll
    for (int it = 0; it < 4; ++it) {  // A: 128 x 64 = 1024 chunks
      int li = it * 256 + tid;
      int r = li >> 3, c = li & 7;
      async_copy16((unsigned)(size_t)&As[buf][r][c * 8],
                   &A[(size_t)(m0 + r) * K + kk + c * 8]);
    }
#pragma unroll
    for (int it = 0; it < 8; ++it) {  // B: 256 x 64 = 2048 chunks
      int li = it * 256 + tid;
      int r = li >> 3, c = li & 7;
      async_copy16((unsigned)(size_t)&Bs[buf][r][c * 8],
                   &BT[(size_t)(n0 + r) * K + kk + c * 8]);
    }
  };

  v8f acc[4][4];
  v8f vzero = {};
#pragma unroll
  for (int i = 0; i < 4; ++i)
#pragma unroll
    for (int j = 0; j < 4; ++j) acc[i][j] = vzero;

  const int nk = K / GBK;
  stage(0, 0);
  for (int ik = 0; ik < nk; ++ik) {
    const int buf = ik & 1;
    if (ik + 1 < nk) {
      stage(buf ^ 1, (ik + 1) * GBK);        // overlap fill of next tile
      if (ik + 2 < nk) {                     // GL2 prefetch 2 tiles ahead
        __builtin_prefetch(&A[(size_t)(m0 + (tid >> 1)) * K + (ik + 2) * GBK + (tid & 1) * 32], 0, 1);
        __builtin_prefetch(&BT[(size_t)(n0 + tid) * K + (ik + 2) * GBK], 0, 1);
      }
      async_wait_12();                       // drain only the older batch
    } else {
      async_wait_all();
    }
    __syncthreads();

#pragma unroll
    for (int ks = 0; ks < 2; ++ks) {
      v16bf af[4], bfr[4];
#pragma unroll
      for (int i = 0; i < 4; ++i)
        af[i] = make_afrag(&As[buf][wm * 64 + i * 16 + l16][ks * 32 + 16 * half]);
#pragma unroll
      for (int j = 0; j < 4; ++j)
        bfr[j] = make_bfrag(&Bs[buf][wn * 64 + j * 16 + l16][ks * 32 + 16 * half]);
#pragma unroll
      for (int i = 0; i < 4; ++i)
#pragma unroll
        for (int j = 0; j < 4; ++j)
          acc[i][j] = wmma_bf16(af[i], bfr[j], acc[i][j]);
    }
    __syncthreads();  // all waves done reading buf before it is refilled
  }

#pragma unroll
  for (int i = 0; i < 4; ++i)
#pragma unroll
    for (int j = 0; j < 4; ++j)
#pragma unroll
      for (int e = 0; e < 8; ++e) {
        int row = m0 + wm * 64 + i * 16 + e + 8 * half;
        int col = n0 + wn * 64 + j * 16 + l16;
        C[(size_t)row * N + col] = acc[i][j][e];
      }
}

// ---------------------------------------------------------------------------
// Per-head RMSNorm + RoPE (mode 0) f32 -> bf16 [B,nh,S,D];
// mode 2: plain convert, transposed d-major layout [B,nh,D,S] (for V).
// One block (128 thr) per (b,s,h) row of the [B,S,nh,D] input.
// ---------------------------------------------------------------------------
__launch_bounds__(128)
__global__ void normrope_kernel(const float* __restrict__ X,
                                const float* __restrict__ w,
                                bf16_t* __restrict__ out, int nheads, int mode) {
  __shared__ float red[4];
  __shared__ float xs[HDIM];
  const int r = blockIdx.x;
  const int d = threadIdx.x;
  const int h = r % nheads;
  const int s = (r / nheads) % S_LEN;
  const int b = r / (nheads * S_LEN);
  float v  = X[(size_t)r * HDIM + d];
  float nv = v;
  if (mode == 0) {
    float sq = v * v;
#pragma unroll
    for (int m = 16; m >= 1; m >>= 1) sq += __shfl_xor(sq, m, 32);
    if ((d & 31) == 0) red[d >> 5] = sq;
    __syncthreads();
    float tot = red[0] + red[1] + red[2] + red[3];
    float rr  = rsqrtf(tot * (1.0f / HDIM) + 1e-5f);
    nv = v * rr * w[d];
    xs[d] = nv;
    __syncthreads();
    int   j    = d & 63;
    float invf = __expf(-9.210340372f * (float)(2 * j) * (1.0f / HDIM));  // theta^-(2j/D)
    float ang  = (float)s * invf;
    float sn, cs;
    __sincosf(ang, &sn, &cs);
    float other = (d < 64) ? -xs[d + 64] : xs[d - 64];
    nv = nv * cs + other * sn;
  }
  if (mode == 2)
    out[(((size_t)b * nheads + h) * HDIM + d) * S_LEN + s] = (bf16_t)nv;
  else
    out[(((size_t)b * nheads + h) * S_LEN + s) * HDIM + d] = (bf16_t)nv;
}

// ---------------------------------------------------------------------------
// Sliding-window causal flash attention, bf16 WMMA, f32 accum.
// Grid: (S/64, NH, B). Block: 128 thr = 4 waves; wave w owns q rows
// [q0+16w, q0+16w+15]. KV in 64-row blocks; D=128.
// V is pre-transposed [B,hk,D,S]; its tile streams in via the async engine
// while QK^T + softmax run, and is only waited on right before P*V.
// ---------------------------------------------------------------------------
#define VSTR 72  // LDS row stride (bf16): 144B, 16B-aligned, padded

__launch_bounds__(128)
__global__ void flash_attn_kernel(const bf16_t* __restrict__ Q,
                                  const bf16_t* __restrict__ Kc,
                                  const bf16_t* __restrict__ Vc,
                                  bf16_t* __restrict__ O) {
  __shared__ bf16_t Vt[HDIM][VSTR];   // [d][kv]
  __shared__ bf16_t Pl[4][16][VSTR];  // per-wave P tile: [wave][m][k]
  const int tid  = threadIdx.x;
  const int lane = tid & 31;
  const int w    = tid >> 5;
  const int half = lane >> 4, l16 = lane & 15;
  const int q0 = blockIdx.x * 64;
  const int h  = blockIdx.y, b = blockIdx.z;
  const int hk = h >> 2;  // GQA: NH/NKV = 4

  const bf16_t* Qbase = Q  + ((size_t)b * NQH  + h ) * S_LEN * HDIM;
  const bf16_t* Kbase = Kc + ((size_t)b * NKVH + hk) * S_LEN * HDIM;
  const bf16_t* Vbase = Vc + ((size_t)b * NKVH + hk) * HDIM * S_LEN;  // d-major

  // resident Q fragments for this wave's 16 rows (D=128 -> 4 k-steps)
  v16bf qf[4];
  const bf16_t* qrow = Qbase + (size_t)(q0 + w * 16 + l16) * HDIM;
#pragma unroll
  for (int ks = 0; ks < 4; ++ks) qf[ks] = make_afrag(qrow + ks * 32 + 16 * half);

  v8f accO[8];
  v8f vzero = {};
#pragma unroll
  for (int d = 0; d < 8; ++d) accO[d] = vzero;
  float m_run[8], l_run[8];
#pragma unroll
  for (int i = 0; i < 8; ++i) { m_run[i] = -1e30f; l_run[i] = 0.0f; }

  int lo = q0 - WIN + 1;
  if (lo < 0) lo = 0;
  const int kb0 = lo & ~63;
  const float scale = 0.08838834764831845f;  // 1/sqrt(128)

  for (int kb = kb0; kb <= q0; kb += 64) {
    // async stage V tile: Vt[d][0..63] <- V[d][kb..kb+63], 8 chunks/thread
#pragma unroll
    for (int j = 0; j < 8; ++j) {
      int li = j * 128 + tid;
      int dd = li >> 3, c = li & 7;
      async_copy16((unsigned)(size_t)&Vt[dd][c * 8],
                   &Vbase[(size_t)dd * S_LEN + kb + c * 8]);
    }

    // S = Q * K^T  (K rows loaded as B-fragments straight from global)
    v8f accS[4];
#pragma unroll
    for (int t = 0; t < 4; ++t) {
      accS[t] = vzero;
      const bf16_t* krow = Kbase + (size_t)(kb + t * 16 + l16) * HDIM;
#pragma unroll
      for (int ks = 0; ks < 4; ++ks) {
        v16bf kf = make_bfrag(krow + ks * 32 + 16 * half);
        accS[t] = wmma_bf16(qf[ks], kf, accS[t]);
      }
    }

    // mask (causal sliding window) + scale, row maxes
    float rmax[8];
#pragma unroll
    for (int i = 0; i < 8; ++i) rmax[i] = -1e30f;
#pragma unroll
    for (int t = 0; t < 4; ++t)
#pragma unroll
      for (int i = 0; i < 8; ++i) {
        int qpos = q0 + w * 16 + i + 8 * half;
        int kpos = kb + t * 16 + l16;
        bool ok  = (kpos <= qpos) && (qpos - kpos < WIN);
        float s  = ok ? accS[t][i] * scale : -1e30f;
        accS[t][i] = s;
        rmax[i] = fmaxf(rmax[i], s);
      }
#pragma unroll
    for (int i = 0; i < 8; ++i)
#pragma unroll
      for (int msk = 8; msk >= 1; msk >>= 1)
        rmax[i] = fmaxf(rmax[i], __shfl_xor(rmax[i], msk, 32));

    float corr[8], rsum[8];
#pragma unroll
    for (int i = 0; i < 8; ++i) {
      float mnew = fmaxf(m_run[i], rmax[i]);
      corr[i] = __expf(m_run[i] - mnew);
      m_run[i] = mnew;
      rsum[i] = 0.0f;
    }
#pragma unroll
    for (int t = 0; t < 4; ++t)
#pragma unroll
      for (int i = 0; i < 8; ++i) {
        float s = accS[t][i];
        float p = (s <= -1e29f) ? 0.0f : __expf(s - m_run[i]);
        accS[t][i] = p;
        rsum[i] += p;
      }
#pragma unroll
    for (int i = 0; i < 8; ++i) {
#pragma unroll
      for (int msk = 8; msk >= 1; msk >>= 1)
        rsum[i] += __shfl_xor(rsum[i], msk, 32);
      l_run[i] = l_run[i] * corr[i] + rsum[i];
    }

    // rescale running O
#pragma unroll
    for (int d = 0; d < 8; ++d)
#pragma unroll
      for (int i = 0; i < 8; ++i) accO[d][i] *= corr[i];

    // P (C-layout) -> LDS -> re-fragment as WMMA-A
#pragma unroll
    for (int t = 0; t < 4; ++t)
#pragma unroll
      for (int i = 0; i < 8; ++i)
        Pl[w][i + 8 * half][t * 16 + l16] = (bf16_t)accS[t][i];
    async_wait_all();   // V tile landed in LDS
    __syncthreads();    // V + P visible to all waves

    // O += P * V   (K-dim 64 -> 2 k-steps, N-dim 128 -> 8 tiles)
#pragma unroll
    for (int ks = 0; ks < 2; ++ks) {
      v16bf pf = make_afrag(&Pl[w][l16][ks * 32 + 16 * half]);
#pragma unroll
      for (int d = 0; d < 8; ++d) {
        v16bf vf = make_bfrag(&Vt[d * 16 + l16][ks * 32 + 16 * half]);
        accO[d] = wmma_bf16(pf, vf, accO[d]);
      }
    }
    __syncthreads();  // protect Vt/Pl before next iteration overwrites
  }

  // normalize and store to [B, S, NH*D] bf16 (ready for out-projection GEMM)
#pragma unroll
  for (int d = 0; d < 8; ++d)
#pragma unroll
    for (int i = 0; i < 8; ++i) {
      int qpos = q0 + w * 16 + i + 8 * half;
      int col  = h * HDIM + d * 16 + l16;
      float val = accO[d][i] / l_run[i];
      O[((size_t)b * S_LEN + qpos) * (NQH * HDIM) + col] = (bf16_t)val;
    }
}

// ---------------------------------------------------------------------------
// Host orchestration
// ---------------------------------------------------------------------------
extern "C" void kernel_launch(void* const* d_in, const int* in_sizes, int n_in,
                              void* d_out, int out_size, void* d_ws, size_t ws_size,
                              hipStream_t stream) {
  (void)in_sizes; (void)n_in; (void)out_size; (void)ws_size;
  const float* hs  = (const float*)d_in[0];
  const float* wq  = (const float*)d_in[1];
  const float* wk  = (const float*)d_in[2];
  const float* wv  = (const float*)d_in[3];
  const float* wo  = (const float*)d_in[4];
  const float* qnw = (const float*)d_in[5];
  const float* knw = (const float*)d_in[6];
  float* out = (float*)d_out;

  const size_t M  = 4096;   // B*S
  const size_t H  = 4096;   // hidden
  const size_t NQ = 4096;   // NH*D
  const size_t NK = 1024;   // NKV*D

  char* ws = (char*)d_ws;
  size_t off = 0;
  auto take = [&](size_t bytes) -> char* {
    char* p = ws + off;
    off = (off + bytes + 255) & ~(size_t)255;
    return p;
  };
  bf16_t* Xb  = (bf16_t*)take(M * H * 2);
  bf16_t* Wqb = (bf16_t*)take(H * NQ * 2);   // transposed: [NQ, H]
  bf16_t* Wkb = (bf16_t*)take(H * NK * 2);   // transposed: [NK, H]
  bf16_t* Wvb = (bf16_t*)take(H * NK * 2);   // transposed: [NK, H]
  bf16_t* Wob = (bf16_t*)take(NQ * H * 2);   // transposed: [H, NQ]
  float*  Qf  = (float*)take(M * NQ * 4);
  float*  Kf  = (float*)take(M * NK * 4);
  float*  Vf  = (float*)take(M * NK * 4);
  bf16_t* Qb  = (bf16_t*)take(M * NQ * 2);   // [B,NH,S,D]
  bf16_t* Kb  = (bf16_t*)take(M * NK * 2);   // [B,NKV,S,D]
  bf16_t* Vb  = (bf16_t*)take(M * NK * 2);   // [B,NKV,D,S] (d-major)
  bf16_t* Ob  = (bf16_t*)take(M * NQ * 2);   // [B,S,NH*D]

  // activations: straight convert; weights: convert + transpose (once)
  cvt_kernel<<<dim3((unsigned)((M * H + 255) / 256)), 256, 0, stream>>>(hs, Xb,
                                                                        (int)(M * H));
  dim3 tb(32, 8);
  cvt_transpose_kernel<<<dim3(NQ / 32, H / 32), tb, 0, stream>>>(wq, Wqb, (int)H, (int)NQ);
  cvt_transpose_kernel<<<dim3(NK / 32, H / 32), tb, 0, stream>>>(wk, Wkb, (int)H, (int)NK);
  cvt_transpose_kernel<<<dim3(NK / 32, H / 32), tb, 0, stream>>>(wv, Wvb, (int)H, (int)NK);
  cvt_transpose_kernel<<<dim3(H / 32, NQ / 32), tb, 0, stream>>>(wo, Wob, (int)NQ, (int)H);

  // QKV projections (bf16 WMMA, f32 out)
  gemm_bf16_kernel<<<dim3(NQ / GBN, M / GBM), 256, 0, stream>>>(Xb, Wqb, Qf,
                                                                (int)M, (int)NQ, (int)H);
  gemm_bf16_kernel<<<dim3(NK / GBN, M / GBM), 256, 0, stream>>>(Xb, Wkb, Kf,
                                                                (int)M, (int)NK, (int)H);
  gemm_bf16_kernel<<<dim3(NK / GBN, M / GBM), 256, 0, stream>>>(Xb, Wvb, Vf,
                                                                (int)M, (int)NK, (int)H);

  // QK RMSNorm + RoPE + relayout; V -> d-major relayout
  normrope_kernel<<<dim3(2 * 2048 * 32), 128, 0, stream>>>(Qf, qnw, Qb, NQH, 0);
  normrope_kernel<<<dim3(2 * 2048 * 8),  128, 0, stream>>>(Kf, knw, Kb, NKVH, 0);
  normrope_kernel<<<dim3(2 * 2048 * 8),  128, 0, stream>>>(Vf, knw, Vb, NKVH, 2);

  // sliding-window flash attention
  flash_attn_kernel<<<dim3(S_LEN / 64, NQH, 2), 128, 0, stream>>>(Qb, Kb, Vb, Ob);

  // output projection -> d_out (f32)
  gemm_bf16_kernel<<<dim3(H / GBN, M / GBM), 256, 0, stream>>>(Ob, Wob, out,
                                                               (int)M, (int)H, (int)NQ);
}
